// MultiTaskLearningDecoder_90185723281998
// MI455X (gfx1250) — matrix operand order repaired and checked by reading, not compiled, not run on hardware
//
#include <hip/hip_runtime.h>
#include <hip/hip_bf16.h>
#include <stdint.h>
#include <stddef.h>

typedef __attribute__((ext_vector_type(16))) _Float16 v16h;
typedef __attribute__((ext_vector_type(8)))  _Float16 v8h;
typedef __attribute__((ext_vector_type(8)))  float    v8f;
typedef __attribute__((ext_vector_type(4)))  float    v4f;

#define B_    16
#define C_    8
#define T_    512
#define N_    128          // B_*C_
#define D_    256
#define H_    256
#define G3_   768          // 3*H_
#define IN_PW  1024
#define IN_PPH 1280
#define IN_IPH 1536
#define TASK_STRIDE 131072 // B_*2*C_*T_

// ---------------------------------------------------------------------------
// Kernel: f32 -> f16 conversion (weights + character inputs)
// ---------------------------------------------------------------------------
__global__ void cvt_f32_to_f16(const float* __restrict__ src,
                               _Float16* __restrict__ dst, int n) {
  int i = blockIdx.x * blockDim.x + threadIdx.x;
  if (i < n) dst[i] = (_Float16)src[i];
}

// ---------------------------------------------------------------------------
// Kernel: precompute time-invariant part of the GRU gate pre-activations.
// gsT[o][n] = bih[o] + (o<2H ? bhh[o] : 0) + sent[n,:]@W[o,256:512]
//                                          + ctx[n,:]@W[o,512:768]
// Stored transposed (o-major, n-minor) -> loads straight into the WMMA f32
// C-layout (8 contiguous floats per lane).
// ---------------------------------------------------------------------------
__global__ void gstatic_kernel(const float* __restrict__ W, int IN,
                               const float* __restrict__ bih,
                               const float* __restrict__ bhh,
                               const float* __restrict__ sent,
                               const float* __restrict__ ctx,
                               float* __restrict__ gsT) {
  int idx = blockIdx.x * blockDim.x + threadIdx.x;
  if (idx >= G3_ * N_) return;
  int o = idx / N_;
  int n = idx % N_;
  const float* wr = W + (size_t)o * IN;
  const float* sv = sent + (size_t)n * D_;
  const float* cv = ctx  + (size_t)n * D_;
  float s = bih[o] + (o < 2 * H_ ? bhh[o] : 0.0f);
  #pragma unroll 4
  for (int k = 0; k < D_; ++k) s += sv[k] * wr[256 + k];
  #pragma unroll 4
  for (int k = 0; k < D_; ++k) s += cv[k] * wr[512 + k];
  gsT[(size_t)o * N_ + n] = s;
}

// ---------------------------------------------------------------------------
// WMMA fragment loaders (wave32, 16x16x32 f16; contiguous K order k = 16h + s, same in A and B)
// ---------------------------------------------------------------------------
// A (16x32 f16): lane m = lane&15; lanes<16 cover K = k..k+15,
// lanes>=16 cover K = k+16..k+31 -> two contiguous 16B loads.
__device__ __forceinline__ v16h load_a_frag(const _Float16* src,
                                            int rowStrideHalves,
                                            int kl, int lane) {
  int m  = lane & 15;
  int k0 = kl + ((lane & 16) ? 16 : 0);
  const _Float16* p = src + (size_t)m * rowStrideHalves + k0;
  union { v16h v; v8h h[2]; } u;
  u.h[0] = *(const v8h*)(p);
  u.h[1] = *(const v8h*)(p + 8);
  return u.v;
}

// B (32x16 f16): out-col = lane&15, lanes<16 cover K = k..k+15, lanes>=16
// K = k+16..k+31. W stored row-major [out][IN] -> 32B contiguous per lane.
__device__ __forceinline__ v16h load_b_frag(const _Float16* W, int IN,
                                            int oBase, int kW, int lane) {
  int o  = oBase + (lane & 15);
  int kk = kW + ((lane & 16) ? 16 : 0);
  return *(const v16h*)(W + (size_t)o * IN + kk);
}

// ---------------------------------------------------------------------------
// One GRU phase: gates = gstatic + [x | h-segments] @ W.T, then the
// zero-h GRU nonlinearity, h written to LDS, linear head written to d_out.
// ---------------------------------------------------------------------------
template <int NSEG>
__device__ __forceinline__ void gru_phase(
    const _Float16* __restrict__ W, int IN,
    const float* __restrict__ gsT, const float* __restrict__ bhh,
    const v16h* xa,                       // 8 preloaded x_t A-fragments
    const _Float16* const* hsegs, const int* segK0,
    _Float16* hout, float* headpart,
    const float* headW,                   // LDS copy, [2][256]
    const float* __restrict__ headB,
    float* __restrict__ outTask, int t, int n0, int tid) {
  const int wave = tid >> 5;
  const int lane = tid & 31;
  const int colb = wave * 16;                // this wave's 16 out-cols per gate
  const int mb   = (lane & 16) ? 8 : 0;      // C/D row-half per lane group
  const int nloc = lane & 15;

  // Seed accumulators from the precomputed static part (contiguous 32B).
  v8f aR = *(const v8f*)(gsT + (size_t)(colb + nloc) * N_        + n0 + mb);
  v8f aZ = *(const v8f*)(gsT + (size_t)(256 + colb + nloc) * N_  + n0 + mb);
  v8f aN = *(const v8f*)(gsT + (size_t)(512 + colb + nloc) * N_  + n0 + mb);

  // K-segment 0: x_t (preloaded A-fragments), weight cols 0..255.
  #pragma unroll
  for (int kt = 0; kt < 8; ++kt) {
    v16h bR = load_b_frag(W, IN, colb,        kt * 32, lane);
    v16h bZ = load_b_frag(W, IN, 256 + colb,  kt * 32, lane);
    v16h bN = load_b_frag(W, IN, 512 + colb,  kt * 32, lane);
    aR = __builtin_amdgcn_wmma_f32_16x16x32_f16(false, xa[kt], false, bR, (short)0, aR, false, false);
    aZ = __builtin_amdgcn_wmma_f32_16x16x32_f16(false, xa[kt], false, bZ, (short)0, aZ, false, false);
    aN = __builtin_amdgcn_wmma_f32_16x16x32_f16(false, xa[kt], false, bN, (short)0, aN, false, false);
  }
  // Hidden-state K-segments from LDS.
  #pragma unroll
  for (int s = 0; s < NSEG; ++s) {
    const _Float16* hs = hsegs[s];
    const int k0 = segK0[s];
    #pragma unroll
    for (int kt = 0; kt < 8; ++kt) {
      v16h a  = load_a_frag(hs, H_, kt * 32, lane);
      int  kW = k0 + kt * 32;
      v16h bR = load_b_frag(W, IN, colb,       kW, lane);
      v16h bZ = load_b_frag(W, IN, 256 + colb, kW, lane);
      v16h bN = load_b_frag(W, IN, 512 + colb, kW, lane);
      aR = __builtin_amdgcn_wmma_f32_16x16x32_f16(false, a, false, bR, (short)0, aR, false, false);
      aZ = __builtin_amdgcn_wmma_f32_16x16x32_f16(false, a, false, bZ, (short)0, aZ, false, false);
      aN = __builtin_amdgcn_wmma_f32_16x16x32_f16(false, a, false, bN, (short)0, aN, false, false);
    }
  }

  // GRU nonlinearity (h_prev == 0): r=sig(aR), z=sig(aZ), n=tanh(aN+r*bhh_n),
  // h=(1-z)*n.  bih + bhh_{r,z} are already folded into gstatic.
  const float bn = bhh[512 + colb + nloc];
  v8f h;
  #pragma unroll
  for (int g = 0; g < 8; ++g) {
    float r  = 1.0f / (1.0f + __expf(-aR[g]));
    float z  = 1.0f / (1.0f + __expf(-aZ[g]));
    float nn = tanhf(aN[g] + r * bn);
    h[g] = (1.0f - z) * nn;
  }

  __syncthreads();  // everyone done reading h_prev from LDS
  const int col = colb + nloc;
  #pragma unroll
  for (int g = 0; g < 8; ++g) hout[(mb + g) * H_ + col] = (_Float16)h[g];
  __syncthreads();  // new h visible to all waves

  // Head: y[n,j] = h[n,:] @ headW[j,:] + headB[j]; 32 dots split 4-ways.
  if (tid < 128) {
    int pair = tid >> 2, q = tid & 3;
    int hn = pair >> 1, j = pair & 1;
    const v8h* hv = (const v8h*)(hout + hn * H_ + q * 64);
    const float* wj = headW + j * H_ + q * 64;
    float s = 0.0f;
    #pragma unroll
    for (int i = 0; i < 8; ++i) {
      v8h hx = hv[i];
      v4f w0 = *(const v4f*)(wj + i * 8);
      v4f w1 = *(const v4f*)(wj + i * 8 + 4);
      s += (float)hx[0] * w0[0] + (float)hx[1] * w0[1] +
           (float)hx[2] * w0[2] + (float)hx[3] * w0[3] +
           (float)hx[4] * w1[0] + (float)hx[5] * w1[1] +
           (float)hx[6] * w1[2] + (float)hx[7] * w1[3];
    }
    headpart[pair * 4 + q] = s;
  }
  __syncthreads();
  if (tid < 32) {
    int hn = tid >> 1, j = tid & 1;
    float s = headpart[tid * 4 + 0] + headpart[tid * 4 + 1] +
              headpart[tid * 4 + 2] + headpart[tid * 4 + 3] + headB[j];
    int ng = n0 + hn;
    int bb = ng >> 3, cc = ng & 7;
    outTask[(((size_t)bb * 2 + j) * C_ + cc) * T_ + t] = s;
  }
}

// ---------------------------------------------------------------------------
// Persistent sequential kernel: 8 blocks x 16 rows, internal loop over T.
// ---------------------------------------------------------------------------
__launch_bounds__(512, 1)
__global__ void gru_seq_kernel(
    const _Float16* __restrict__ chars16,
    const _Float16* __restrict__ Wpw, const _Float16* __restrict__ Wpph,
    const _Float16* __restrict__ Wiph,
    const float* __restrict__ gsPW, const float* __restrict__ gsPPH,
    const float* __restrict__ gsIPH,
    const float* __restrict__ bhhPW, const float* __restrict__ bhhPPH,
    const float* __restrict__ bhhIPH,
    const float* __restrict__ WhdPW, const float* __restrict__ bhdPW,
    const float* __restrict__ WhdPPH, const float* __restrict__ bhdPPH,
    const float* __restrict__ WhdIPH, const float* __restrict__ bhdIPH,
    float* __restrict__ out) {
  __shared__ _Float16 hpw[16 * H_];
  __shared__ _Float16 hpph[16 * H_];
  __shared__ _Float16 hiph[16 * H_];
  __shared__ float headpart[32 * 4];
  __shared__ float hwLDS[3 * 2 * H_];   // head weights staged in LDS

  const int tid  = threadIdx.x;
  const int lane = tid & 31;
  const int n0   = blockIdx.x * 16;

  for (int i = tid; i < 16 * H_; i += 512) {
    hpw[i] = (_Float16)0.0f; hpph[i] = (_Float16)0.0f; hiph[i] = (_Float16)0.0f;
  }
  for (int i = tid; i < 2 * H_; i += 512) {
    hwLDS[i]            = WhdPW[i];
    hwLDS[2 * H_ + i]   = WhdPPH[i];
    hwLDS[4 * H_ + i]   = WhdIPH[i];
  }
  __syncthreads();

  const _Float16* segs1[1] = { hpw };
  const int       k01[1]   = { 768 };
  const _Float16* segs2[2] = { hpw, hpph };
  const int       k02[2]   = { 768, 1024 };
  const _Float16* segs3[3] = { hpw, hpph, hiph };
  const int       k03[3]   = { 768, 1024, 1280 };

  for (int t = 0; t < T_; ++t) {
    // Break cross-iteration CSE/LICM of the (loop-invariant-address) weight
    // loads: without this the compiler hoists all B-fragments out of the t
    // loop and spills ~7KB/lane to scratch, replacing shared L2-resident
    // weight streams with per-wave private scratch traffic.
    asm volatile("" ::: "memory");

    const _Float16* xrow = chars16 + ((size_t)n0 * T_ + t) * D_;

    // Warm next step's x tile into cache while this step computes.
    if (t + 1 < T_ && tid < 16) {
      __builtin_prefetch(chars16 + ((size_t)(n0 + tid) * T_ + (t + 1)) * D_, 0, 1);
    }

    // Preload the x_t A-fragments once; shared by all three GRU phases.
    v16h xa[8];
    #pragma unroll
    for (int kt = 0; kt < 8; ++kt) xa[kt] = load_a_frag(xrow, T_ * D_, kt * 32, lane);

    gru_phase<1>(Wpw,  IN_PW,  gsPW,  bhhPW,  xa, segs1, k01, hpw,
                 headpart, hwLDS,          bhdPW,  out + 0 * TASK_STRIDE, t, n0, tid);
    gru_phase<2>(Wpph, IN_PPH, gsPPH, bhhPPH, xa, segs2, k02, hpph,
                 headpart, hwLDS + 2 * H_, bhdPPH, out + 1 * TASK_STRIDE, t, n0, tid);
    gru_phase<3>(Wiph, IN_IPH, gsIPH, bhhIPH, xa, segs3, k03, hiph,
                 headpart, hwLDS + 4 * H_, bhdIPH, out + 2 * TASK_STRIDE, t, n0, tid);
  }
}

// ---------------------------------------------------------------------------
extern "C" void kernel_launch(void* const* d_in, const int* in_sizes, int n_in,
                              void* d_out, int out_size, void* d_ws, size_t ws_size,
                              hipStream_t stream) {
  (void)in_sizes; (void)n_in; (void)out_size; (void)ws_size;

  const float* chars  = (const float*)d_in[0];
  const float* sent   = (const float*)d_in[1];
  const float* ctx    = (const float*)d_in[2];
  const float* pw_Wih = (const float*)d_in[3];
  const float* pw_bih = (const float*)d_in[4];
  const float* pw_bhh = (const float*)d_in[5];
  const float* pph_Wih= (const float*)d_in[6];
  const float* pph_bih= (const float*)d_in[7];
  const float* pph_bhh= (const float*)d_in[8];
  const float* iph_Wih= (const float*)d_in[9];
  const float* iph_bih= (const float*)d_in[10];
  const float* iph_bhh= (const float*)d_in[11];
  const float* pw_W   = (const float*)d_in[12];
  const float* pw_b   = (const float*)d_in[13];
  const float* pph_W  = (const float*)d_in[14];
  const float* pph_b  = (const float*)d_in[15];
  const float* iph_W  = (const float*)d_in[16];
  const float* iph_b  = (const float*)d_in[17];
  float* out = (float*)d_out;

  // Workspace layout (all chunks 256B-aligned by construction).
  char* ws = (char*)d_ws;
  _Float16* wpw16  = (_Float16*)ws;  ws += (size_t)G3_ * IN_PW  * 2;  // 1.5 MB
  _Float16* wpph16 = (_Float16*)ws;  ws += (size_t)G3_ * IN_PPH * 2;  // 1.9 MB
  _Float16* wiph16 = (_Float16*)ws;  ws += (size_t)G3_ * IN_IPH * 2;  // 2.3 MB
  _Float16* chars16= (_Float16*)ws;  ws += (size_t)N_ * T_ * D_ * 2;  // 33.6 MB
  float* gsPW  = (float*)ws;         ws += (size_t)G3_ * N_ * 4;
  float* gsPPH = (float*)ws;         ws += (size_t)G3_ * N_ * 4;
  float* gsIPH = (float*)ws;         ws += (size_t)G3_ * N_ * 4;

  // 1) f16 conversions (weights + inputs).
  {
    int n;
    n = G3_ * IN_PW;
    cvt_f32_to_f16<<<(n + 255) / 256, 256, 0, stream>>>(pw_Wih, wpw16, n);
    n = G3_ * IN_PPH;
    cvt_f32_to_f16<<<(n + 255) / 256, 256, 0, stream>>>(pph_Wih, wpph16, n);
    n = G3_ * IN_IPH;
    cvt_f32_to_f16<<<(n + 255) / 256, 256, 0, stream>>>(iph_Wih, wiph16, n);
    n = N_ * T_ * D_;
    cvt_f32_to_f16<<<(n + 255) / 256, 256, 0, stream>>>(chars, chars16, n);
  }

  // 2) time-invariant gate pre-activations (uses original f32 weights).
  {
    int n = G3_ * N_;
    int g = (n + 255) / 256;
    gstatic_kernel<<<g, 256, 0, stream>>>(pw_Wih,  IN_PW,  pw_bih,  pw_bhh,  sent, ctx, gsPW);
    gstatic_kernel<<<g, 256, 0, stream>>>(pph_Wih, IN_PPH, pph_bih, pph_bhh, sent, ctx, gsPPH);
    gstatic_kernel<<<g, 256, 0, stream>>>(iph_Wih, IN_IPH, iph_bih, iph_bhh, sent, ctx, gsIPH);
  }

  // 3) persistent sequential GRU chain: 8 blocks x 512 threads (16 waves).
  gru_seq_kernel<<<8, 512, 0, stream>>>(
      chars16, wpw16, wpph16, wiph16,
      gsPW, gsPPH, gsIPH,
      pw_bhh, pph_bhh, iph_bhh,
      pw_W, pw_b, pph_W, pph_b, iph_W, iph_b,
      out);
}
